// GNNCentroid_19628000542960
// MI455X (gfx1250) — matrix operand order repaired and checked by reading, not compiled, not run on hardware
//
#include <hip/hip_runtime.h>
#include <hip/hip_bf16.h>
#include <stdint.h>

// ---------------------------------------------------------------- constants
#define NND   65536          // nodes
#define NED   1048576        // edges
#define HID   256
#define ACTD  5
#define LDA   288            // padded feature stride (covers 261, multiple of 32)

typedef __attribute__((ext_vector_type(16))) __bf16 v16bf;
typedef __attribute__((ext_vector_type(8)))  __bf16 v8bf;
typedef __attribute__((ext_vector_type(4)))  __bf16 v4bf;
typedef __attribute__((ext_vector_type(8)))  float  v8f;

// fp32 -> bf16, round-to-nearest-even via integer trick
static __device__ __forceinline__ __bf16 f2bf(float f) {
    union { float f; unsigned u; } v; v.f = f;
    unsigned r = v.u + 0x7FFFu + ((v.u >> 16) & 1u);
    unsigned short h = (unsigned short)(r >> 16);
    return __builtin_bit_cast(__bf16, h);
}

static __device__ __forceinline__ void atomAddF(float* p, float v) {
    __hip_atomic_fetch_add(p, v, __ATOMIC_RELAXED, __HIP_MEMORY_SCOPE_AGENT);
}

// ---------------------------------------------------------------- degree / norm
__global__ __launch_bounds__(256) void k_count_deg(const long long* __restrict__ ei,
                                                   unsigned* __restrict__ deg) {
    int e = blockIdx.x * blockDim.x + threadIdx.x;
    if (e >= NED) return;
    unsigned d = (unsigned)ei[NED + e];
    atomicAdd(&deg[d], 1u);
}

__global__ __launch_bounds__(256) void k_dinv(const unsigned* __restrict__ deg,
                                              float* __restrict__ dinv) {
    int i = blockIdx.x * blockDim.x + threadIdx.x;
    if (i >= NND) return;
    dinv[i] = rsqrtf((float)deg[i] + 1.0f);   // +1 self loop
}

// ---------------------------------------------------------------- weight packing
// Pack row-major fp32 W[Krows x Ncols] into bf16 B-fragments for
// v_wmma_f32_16x16x32_bf16. Fragment (kt,nt): 32 lanes x 16 bf16 contiguous.
// lane<16 : column nt*16+lane, K = kt*32 + j      (j = 0..15)
// lane>=16: column nt*16+(lane-16), K = kt*32+16+j
__global__ __launch_bounds__(256) void k_pack_w(const float* __restrict__ W,
                                                __bf16* __restrict__ out,
                                                int Krows, int Kpad, int Ncols) {
    int idx = blockIdx.x * blockDim.x + threadIdx.x;
    int total = Kpad * Ncols;
    if (idx >= total) return;
    int frag = idx >> 9;          // /512
    int r    = idx & 511;
    int lane = r >> 4;
    int j    = r & 15;
    int NT   = Ncols >> 4;
    int kt   = frag / NT;
    int nt   = frag - kt * NT;
    int kk   = kt * 32 + ((lane >> 4) << 4) + j;
    int col  = (nt << 4) + (lane & 15);
    float v  = (kk < Krows) ? W[(size_t)kk * Ncols + col] : 0.0f;
    out[idx] = f2bf(v);
}

// ---------------------------------------------------------------- WMMA GEMM
// C[M x 256] = Ah[M x K](bf16, stride LDA) * Bp(packed bf16), C fp32 stride LDA.
// Block: 256 thr = 8 waves; wave computes 16x64; block tile 128x64.
// grid = (256/64, M/128)
__global__ __launch_bounds__(256) void k_gemm_wmma(const __bf16* __restrict__ Ah,
                                                   const __bf16* __restrict__ Bp,
                                                   float* __restrict__ C,
                                                   int KT) {
    const int lane = threadIdx.x & 31;
    const int wv   = threadIdx.x >> 5;
    const int rowbase = blockIdx.y * 128 + wv * 16;
    const int cbase   = blockIdx.x * 64;
    const int arow = rowbase + (lane & 15);
    const int koff = (lane >> 4) << 4;              // 0 or 16
    const int NT   = 16;                            // 256/16

    v8f acc0 = {}, acc1 = {}, acc2 = {}, acc3 = {};

    for (int kt = 0; kt < KT; ++kt) {
        // ---- A fragment: contiguous k order (K = kt*32 + koff + s), same as packed B; direct bf16 loads (2x 16B)
        const __bf16* pa = Ah + (size_t)arow * LDA + kt * 32 + koff;
        v8bf a0 = *(const v8bf*)(pa);        // K = kbase .. kbase+7
        v8bf a1 = *(const v8bf*)(pa + 8);    // K = kbase+8 .. kbase+15
        v16bf a = __builtin_shufflevector(a0, a1,
                     0, 1, 2, 3, 4, 5, 6, 7, 8, 9, 10, 11, 12, 13, 14, 15);

        // ---- B fragments: contiguous 32B per lane, pre-packed
        const __bf16* pb = Bp + (((size_t)kt * NT + (blockIdx.x * 4)) * 32 + lane) * 16;
        v16bf b0 = *(const v16bf*)(pb);
        v16bf b1 = *(const v16bf*)(pb + 512);
        v16bf b2 = *(const v16bf*)(pb + 1024);
        v16bf b3 = *(const v16bf*)(pb + 1536);

        acc0 = __builtin_amdgcn_wmma_f32_16x16x32_bf16(false, a, false, b0, (short)0, acc0, false, false);
        acc1 = __builtin_amdgcn_wmma_f32_16x16x32_bf16(false, a, false, b1, (short)0, acc1, false, false);
        acc2 = __builtin_amdgcn_wmma_f32_16x16x32_bf16(false, a, false, b2, (short)0, acc2, false, false);
        acc3 = __builtin_amdgcn_wmma_f32_16x16x32_bf16(false, a, false, b3, (short)0, acc3, false, false);
    }

    // ---- C/D layout: VGPR r -> M = r (lanes 0-15) / r+8 (lanes 16-31), N = lane&15
    const int crow0 = rowbase + ((lane >> 4) << 3);
    const int ccol  = cbase + (lane & 15);
#pragma unroll
    for (int r = 0; r < 8; ++r) {
        float* co = C + (size_t)(crow0 + r) * LDA + ccol;
        co[0]  = acc0[r];
        co[16] = acc1[r];
        co[32] = acc2[r];
        co[48] = acc3[r];
    }
}

// ---------------------------------------------------------------- small GEMMs
// H[N x 256] = X[N x 3] * W1[3 x 256]
__global__ __launch_bounds__(256) void k_gemm_in3(const float* __restrict__ x,
                                                  const float* __restrict__ W1,
                                                  float* __restrict__ H) {
    int idx = blockIdx.x * blockDim.x + threadIdx.x;
    int node = idx >> 6;
    int c    = (idx & 63) << 2;
    if (node >= NND) return;
    float x0 = x[node * 3 + 0], x1 = x[node * 3 + 1], x2 = x[node * 3 + 2];
    float4 w0 = *(const float4*)(W1 + c);
    float4 w1 = *(const float4*)(W1 + 256 + c);
    float4 w2 = *(const float4*)(W1 + 512 + c);
    float4 h;
    h.x = x0 * w0.x + x1 * w1.x + x2 * w2.x;
    h.y = x0 * w0.y + x1 * w1.y + x2 * w2.y;
    h.z = x0 * w0.z + x1 * w1.z + x2 * w2.z;
    h.w = x0 * w0.w + x1 * w1.w + x2 * w2.w;
    *(float4*)(H + (size_t)node * LDA + c) = h;
}

// H6[N x 3] = A[N x 256(stride LDA)] * W6[256 x 3]
__global__ __launch_bounds__(256) void k_gemm_out3(const float* __restrict__ A,
                                                   const float* __restrict__ W6,
                                                   float* __restrict__ H6) {
    int node = blockIdx.x * blockDim.x + threadIdx.x;
    if (node >= NND) return;
    const float* a = A + (size_t)node * LDA;
    float s0 = 0.f, s1 = 0.f, s2 = 0.f;
#pragma unroll 4
    for (int k = 0; k < 256; k += 4) {
        float4 av = *(const float4*)(a + k);
        s0 += av.x * W6[(k + 0) * 3 + 0] + av.y * W6[(k + 1) * 3 + 0]
            + av.z * W6[(k + 2) * 3 + 0] + av.w * W6[(k + 3) * 3 + 0];
        s1 += av.x * W6[(k + 0) * 3 + 1] + av.y * W6[(k + 1) * 3 + 1]
            + av.z * W6[(k + 2) * 3 + 1] + av.w * W6[(k + 3) * 3 + 1];
        s2 += av.x * W6[(k + 0) * 3 + 2] + av.y * W6[(k + 1) * 3 + 2]
            + av.z * W6[(k + 2) * 3 + 2] + av.w * W6[(k + 3) * 3 + 2];
    }
    H6[node * 3 + 0] = s0;
    H6[node * 3 + 1] = s1;
    H6[node * 3 + 2] = s2;
}

// ---------------------------------------------------------------- aggregation (F=256)
// out = bias + dinv^2 * h   (self loop + bias, full overwrite)
__global__ __launch_bounds__(256) void k_agg_init_vec(float* __restrict__ out,
                                                      const float* __restrict__ h,
                                                      const float* __restrict__ bias,
                                                      const float* __restrict__ dinv) {
    int idx = blockIdx.x * blockDim.x + threadIdx.x;
    int node = idx >> 6;
    int f    = (idx & 63) << 2;
    if (node >= NND) return;
    float di = dinv[node];
    float sl = di * di;
    float4 hv = *(const float4*)(h + (size_t)node * LDA + f);
    float4 bv = *(const float4*)(bias + f);
    float4 o;
    o.x = bv.x + sl * hv.x; o.y = bv.y + sl * hv.y;
    o.z = bv.z + sl * hv.z; o.w = bv.w + sl * hv.w;
    *(float4*)(out + (size_t)node * LDA + f) = o;
}

// scatter-add over edges: 64 threads/edge, 4 feats each; fp32 hardware atomics
__global__ __launch_bounds__(256) void k_agg_edges_vec(float* __restrict__ out,
                                                       const float* __restrict__ h,
                                                       const long long* __restrict__ ei,
                                                       const float* __restrict__ dinv) {
    long long t = (long long)blockIdx.x * blockDim.x + threadIdx.x;
    int e = (int)(t >> 6);
    int f = (int)(t & 63) << 2;
    if (e >= NED) return;
    long long s = ei[e];
    long long d = ei[NED + e];
    float nrm = dinv[s] * dinv[d];
    float4 hv = *(const float4*)(h + (size_t)s * LDA + f);
    float* o = out + (size_t)d * LDA + f;
    atomAddF(o + 0, nrm * hv.x);
    atomAddF(o + 1, nrm * hv.y);
    atomAddF(o + 2, nrm * hv.z);
    atomAddF(o + 3, nrm * hv.w);
}

// ReLU in place (fp32) + emit bf16 mirror for the next WMMA GEMM's A-operand
__global__ __launch_bounds__(256) void k_relu_vec(float* __restrict__ out,
                                                  __bf16* __restrict__ outh) {
    int idx = blockIdx.x * blockDim.x + threadIdx.x;
    int node = idx >> 6;
    int f    = (idx & 63) << 2;
    if (node >= NND) return;
    float4* p = (float4*)(out + (size_t)node * LDA + f);
    float4 v = *p;
    v.x = fmaxf(v.x, 0.f); v.y = fmaxf(v.y, 0.f);
    v.z = fmaxf(v.z, 0.f); v.w = fmaxf(v.w, 0.f);
    *p = v;
    v4bf hv;
    hv[0] = f2bf(v.x); hv[1] = f2bf(v.y); hv[2] = f2bf(v.z); hv[3] = f2bf(v.w);
    *(v4bf*)(outh + (size_t)node * LDA + f) = hv;
}

// cols 256..287: action broadcast (256..260), zeros (261..287); fp32 + bf16 mirror
__global__ __launch_bounds__(256) void k_concat_action(float* __restrict__ bufA,
                                                       __bf16* __restrict__ bufAh,
                                                       const float* __restrict__ action) {
    int idx = blockIdx.x * blockDim.x + threadIdx.x;
    int node = idx >> 5;
    int c    = 256 + (idx & 31);
    if (node >= NND) return;
    float v = 0.f;
    if (c < 256 + ACTD) v = action[(node >> 6) * ACTD + (c - 256)];
    bufA[(size_t)node * LDA + c]  = v;
    bufAh[(size_t)node * LDA + c] = f2bf(v);
}

// ---------------------------------------------------------------- aggregation (F=3)
__global__ __launch_bounds__(256) void k_agg_init_s(float* __restrict__ out,
                                                    const float* __restrict__ h,
                                                    const float* __restrict__ bias,
                                                    const float* __restrict__ dinv) {
    int node = blockIdx.x * blockDim.x + threadIdx.x;
    if (node >= NND) return;
    float di = dinv[node];
    float sl = di * di;
#pragma unroll
    for (int c = 0; c < 3; ++c)
        out[node * 3 + c] = bias[c] + sl * h[node * 3 + c];
}

__global__ __launch_bounds__(256) void k_agg_edges_s(float* __restrict__ out,
                                                     const float* __restrict__ h,
                                                     const long long* __restrict__ ei,
                                                     const float* __restrict__ dinv) {
    int e = blockIdx.x * blockDim.x + threadIdx.x;
    if (e >= NED) return;
    long long s = ei[e];
    long long d = ei[NED + e];
    float nrm = dinv[s] * dinv[d];
#pragma unroll
    for (int c = 0; c < 3; ++c)
        atomAddF(out + d * 3 + c, nrm * h[s * 3 + c]);
}

__global__ __launch_bounds__(256) void k_relu_s(float* __restrict__ out) {
    int i = blockIdx.x * blockDim.x + threadIdx.x;
    if (i >= NND * 3) return;
    out[i] = fmaxf(out[i], 0.f);
}

// ---------------------------------------------------------------- launcher
extern "C" void kernel_launch(void* const* d_in, const int* in_sizes, int n_in,
                              void* d_out, int out_size, void* d_ws, size_t ws_size,
                              hipStream_t stream) {
    const float*     x      = (const float*)d_in[0];
    const float*     action = (const float*)d_in[1];
    const long long* ei     = (const long long*)d_in[2];
    const float* W1 = (const float*)d_in[3];  const float* b1 = (const float*)d_in[4];
    const float* W2 = (const float*)d_in[5];  const float* b2 = (const float*)d_in[6];
    const float* W3 = (const float*)d_in[7];  const float* b3 = (const float*)d_in[8];
    const float* W4 = (const float*)d_in[9];  const float* b4 = (const float*)d_in[10];
    const float* W5 = (const float*)d_in[11]; const float* b5 = (const float*)d_in[12];
    const float* W6 = (const float*)d_in[13]; const float* b6 = (const float*)d_in[14];
    float* out = (float*)d_out;

    // workspace carve-up (256B aligned chunks)
    char* ws = (char*)d_ws;
    size_t off = 0;
    auto alloc = [&](size_t bytes) -> void* {
        off = (off + 255) & ~(size_t)255;
        void* p = ws + off;
        off += bytes;
        return p;
    };
    unsigned* deg   = (unsigned*)alloc((size_t)NND * 4);
    float*    dinv  = (float*)   alloc((size_t)NND * 4);
    __bf16*   wp2   = (__bf16*)  alloc((size_t)8 * 16 * 512 * 2);   // 256x256
    __bf16*   wp3   = (__bf16*)  alloc((size_t)8 * 16 * 512 * 2);
    __bf16*   wp5   = (__bf16*)  alloc((size_t)8 * 16 * 512 * 2);
    __bf16*   wp4   = (__bf16*)  alloc((size_t)9 * 16 * 512 * 2);   // 288x256 (261 padded)
    float*    bufA  = (float*)   alloc((size_t)NND * LDA * 4);      // fp32 activations
    float*    bufB  = (float*)   alloc((size_t)NND * LDA * 4);      // GEMM output / messages
    __bf16*   bufAh = (__bf16*)  alloc((size_t)NND * LDA * 2);      // bf16 mirror of bufA
    (void)ws_size; (void)in_sizes; (void)n_in; (void)out_size;

    const int B = 256;
    const int gNode64 = (NND * 64) / B;       // node x 64-thread kernels
    const int gEdge64 = (int)(((long long)NED * 64) / B);
    const dim3 gWmma(256 / 64, NND / 128);    // (4, 512)

    // --- degrees / norms
    hipMemsetAsync(deg, 0, (size_t)NND * 4, stream);
    k_count_deg<<<NED / B, B, 0, stream>>>(ei, deg);
    k_dinv<<<NND / B, B, 0, stream>>>(deg, dinv);

    // --- pack weights to bf16 B-fragments
    k_pack_w<<<(256 * 256) / B, B, 0, stream>>>(W2, wp2, 256, 256, 256);
    k_pack_w<<<(256 * 256) / B, B, 0, stream>>>(W3, wp3, 256, 256, 256);
    k_pack_w<<<(256 * 256) / B, B, 0, stream>>>(W5, wp5, 256, 256, 256);
    k_pack_w<<<(288 * 256) / B, B, 0, stream>>>(W4, wp4, 261, 288, 256);

    // --- layer 1: 3 -> 256
    k_gemm_in3<<<gNode64, B, 0, stream>>>(x, W1, bufB);
    k_agg_init_vec<<<gNode64, B, 0, stream>>>(bufA, bufB, b1, dinv);
    k_agg_edges_vec<<<gEdge64, B, 0, stream>>>(bufA, bufB, ei, dinv);
    k_relu_vec<<<gNode64, B, 0, stream>>>(bufA, bufAh);

    // --- layer 2
    k_gemm_wmma<<<gWmma, B, 0, stream>>>(bufAh, wp2, bufB, 8);
    k_agg_init_vec<<<gNode64, B, 0, stream>>>(bufA, bufB, b2, dinv);
    k_agg_edges_vec<<<gEdge64, B, 0, stream>>>(bufA, bufB, ei, dinv);
    k_relu_vec<<<gNode64, B, 0, stream>>>(bufA, bufAh);

    // --- layer 3
    k_gemm_wmma<<<gWmma, B, 0, stream>>>(bufAh, wp3, bufB, 8);
    k_agg_init_vec<<<gNode64, B, 0, stream>>>(bufA, bufB, b3, dinv);
    k_agg_edges_vec<<<gEdge64, B, 0, stream>>>(bufA, bufB, ei, dinv);
    k_relu_vec<<<gNode64, B, 0, stream>>>(bufA, bufAh);

    // --- concat action into cols 256..287, then layer 4: 261(pad 288) -> 256
    k_concat_action<<<(NND * 32) / B, B, 0, stream>>>(bufA, bufAh, action);
    k_gemm_wmma<<<gWmma, B, 0, stream>>>(bufAh, wp4, bufB, 9);
    k_agg_init_vec<<<gNode64, B, 0, stream>>>(bufA, bufB, b4, dinv);
    k_agg_edges_vec<<<gEdge64, B, 0, stream>>>(bufA, bufB, ei, dinv);
    k_relu_vec<<<gNode64, B, 0, stream>>>(bufA, bufAh);

    // --- layer 5
    k_gemm_wmma<<<gWmma, B, 0, stream>>>(bufAh, wp5, bufB, 8);
    k_agg_init_vec<<<gNode64, B, 0, stream>>>(bufA, bufB, b5, dinv);
    k_agg_edges_vec<<<gEdge64, B, 0, stream>>>(bufA, bufB, ei, dinv);
    k_relu_vec<<<gNode64, B, 0, stream>>>(bufA, bufAh);

    // --- layer 6: 256 -> 3, output to d_out
    k_gemm_out3<<<NND / B, B, 0, stream>>>(bufA, W6, bufB);  // bufB holds H6 (stride 3)
    k_agg_init_s<<<NND / B, B, 0, stream>>>(out, bufB, b6, dinv);
    k_agg_edges_s<<<NED / B, B, 0, stream>>>(out, bufB, ei, dinv);
    k_relu_s<<<(NND * 3 + B - 1) / B, B, 0, stream>>>(out);
}
